// UnsupervisedMPNN_47845935677653
// MI455X (gfx1250) — matrix operand order repaired and checked by reading, not compiled, not run on hardware
//
#include <hip/hip_runtime.h>
#include <cmath>

#define NN   100000
#define EE   200000
#define BBG  1024
#define DDIM 32
#define MAXFQ 8

typedef __bf16 bf16_t;
typedef __attribute__((ext_vector_type(16))) __bf16 v16bf;
typedef __attribute__((ext_vector_type(8)))  float  v8f;

__device__ __forceinline__ v16bf splat16(bf16_t x) {
  v16bf v;
#pragma unroll
  for (int i = 0; i < 16; ++i) v[i] = x;
  return v;
}

__device__ __forceinline__ float sigf(float x) { return 1.0f / (1.0f + expf(-x)); }

// ---------------------------------------------------------------------------
// Node featurize + lin0 + relu : out (N,32)
// feat = [pos_u(16) | pos_d(16) | node_emb[clip(nf)](32) | seed | nf/8]
__global__ void lin0_kernel(const float* __restrict__ pu, const float* __restrict__ pd,
                            const float* __restrict__ seed, const float* __restrict__ nemb,
                            const float* __restrict__ w, const float* __restrict__ b,
                            const int* __restrict__ nfreq, float* __restrict__ out) {
  int idx = blockIdx.x * blockDim.x + threadIdx.x;
  if (idx >= NN * 32) return;
  int n = idx >> 5, d = idx & 31;
  float acc = b[d];
  const float* pur = pu + (long)n * 16;
  const float* pdr = pd + (long)n * 16;
#pragma unroll
  for (int i = 0; i < 16; ++i) acc += pur[i] * w[i * 32 + d];
#pragma unroll
  for (int i = 0; i < 16; ++i) acc += pdr[i] * w[(16 + i) * 32 + d];
  int nf = nfreq[n];
  int c = nf < 0 ? 0 : (nf > MAXFQ ? MAXFQ : nf);
  const float* er = nemb + c * 32;
#pragma unroll
  for (int i = 0; i < 32; ++i) acc += er[i] * w[(32 + i) * 32 + d];
  acc += seed[n] * w[64 * 32 + d];
  acc += ((float)nf * 0.125f) * w[65 * 32 + d];
  out[idx] = fmaxf(acc, 0.0f);
}

// ---------------------------------------------------------------------------
// Edge hidden: h_e = relu([edge_emb | ef/8] @ en_w1 + b1) : (E,32), once
__global__ void edge_hidden_kernel(const float* __restrict__ eemb, const float* __restrict__ w1,
                                   const float* __restrict__ b1, const int* __restrict__ efreq,
                                   float* __restrict__ he) {
  int idx = blockIdx.x * blockDim.x + threadIdx.x;
  if (idx >= EE * 32) return;
  int e = idx >> 5, d = idx & 31;
  int ef = efreq[e];
  int c = ef < 0 ? 0 : (ef > MAXFQ ? MAXFQ : ef);
  const float* er = eemb + c * 32;
  float acc = b1[d];
#pragma unroll
  for (int i = 0; i < 32; ++i) acc += er[i] * w1[i * 32 + d];
  acc += ((float)ef * 0.125f) * w1[32 * 32 + d];
  he[idx] = fmaxf(acc, 0.0f);
}

// ---------------------------------------------------------------------------
__global__ void agg_init_kernel(float* __restrict__ agg, const float* __restrict__ conv_b) {
  int idx = blockIdx.x * blockDim.x + threadIdx.x;
  if (idx < NN * 32) agg[idx] = conv_b[idx & 31];
}

// ---------------------------------------------------------------------------
// Fused per-edge message GEMM using WMMA bf16->f32:
//   msg[e,o] = sum_{k,i} h_e[k]*x[src[e],i]*W2[k, i*32+o] + sum_i x_i*b2[i*32+o]
// GEMM view: P(E x 1024) @ W2f(1024 x 32); A k-step s fragment = x_frag * h_e[s].
// Workgroup = 4 waves; each wave: 64 edges = 4 M-tiles x 2 N-tiles accumulators.
__global__ __launch_bounds__(128) void mp_edge_kernel(
    const float* __restrict__ xin, const float* __restrict__ he,
    const float* __restrict__ w2, const float* __restrict__ b2,
    const int* __restrict__ src, const int* __restrict__ dst,
    float* __restrict__ agg) {
  // B-fragments of W2 pre-swizzled: frag f = 2*s_local + n_tile; per lane 16 bf16.
  __shared__ __attribute__((aligned(32))) bf16_t w2s[16][32][16];  // 16 KB (8 k-steps/chunk)
  __shared__ __attribute__((aligned(32))) bf16_t b2s[2][32][16];   //  2 KB
  __shared__ bf16_t xs[4][64][32];                                  // 16 KB
  __shared__ bf16_t hsh[4][64][32];                                 // 16 KB
  __shared__ int dsts[4][64];                                       //  1 KB

  const int tid  = threadIdx.x;
  const int wave = tid >> 5;
  const int lane = tid & 31;
  const int m    = lane & 15;
  const bool hihalf = lane >= 16;
  const int ebase = blockIdx.x * 256 + wave * 64;

  // Stage x = out[src[e]] and h_e rows (as bf16) and dst for this wave's 64 edges.
  for (int i = lane; i < 64 * 8; i += 32) {
    const int r  = i >> 3;
    const int c4 = (i & 7) << 2;
    const int eg = ebase + r;
    float4 xv = {0.f, 0.f, 0.f, 0.f};
    float4 hv = {0.f, 0.f, 0.f, 0.f};
    if (eg < EE) {
      const int sn = src[eg];
      xv = *(const float4*)(xin + (long)sn * 32 + c4);
      hv = *(const float4*)(he + (long)eg * 32 + c4);
    }
    xs[wave][r][c4 + 0] = (bf16_t)xv.x;  xs[wave][r][c4 + 1] = (bf16_t)xv.y;
    xs[wave][r][c4 + 2] = (bf16_t)xv.z;  xs[wave][r][c4 + 3] = (bf16_t)xv.w;
    hsh[wave][r][c4 + 0] = (bf16_t)hv.x; hsh[wave][r][c4 + 1] = (bf16_t)hv.y;
    hsh[wave][r][c4 + 2] = (bf16_t)hv.z; hsh[wave][r][c4 + 3] = (bf16_t)hv.w;
  }
  for (int r = lane; r < 64; r += 32) {
    const int eg = ebase + r;
    dsts[wave][r] = (eg < EE) ? dst[eg] : 0;
  }
  // b2 fragments (bias "extra k-step"): B[i][o] = b2[i*32+o]
  for (int i = tid; i < 2 * 32 * 16; i += 128) {
    const int t = i >> 9, L = (i >> 4) & 31, e2 = i & 15;
    const int kl = e2 + ((L >= 16) ? 16 : 0);
    const int o  = t * 16 + (L & 15);
    b2s[t][L][e2] = (bf16_t)b2[kl * 32 + o];
  }
  __syncthreads();

  // x part of A fragment (fixed across k-steps). Contiguous K order matching B:
  // lane<16: elem j -> K = j;  lane>=16: K = j+16.
  v16bf xf[4];
#pragma unroll
  for (int T = 0; T < 4; ++T) {
    v16bf v;
#pragma unroll
    for (int e2 = 0; e2 < 16; ++e2) {
      const int k = e2 + (hihalf ? 16 : 0);
      v[e2] = xs[wave][T * 16 + m][k];
    }
    xf[T] = v;
  }

  v8f zero = {};
  v8f acc[4][2];
#pragma unroll
  for (int T = 0; T < 4; ++T) { acc[T][0] = zero; acc[T][1] = zero; }

  for (int chunk = 0; chunk < 4; ++chunk) {
    __syncthreads();
    // Fill swizzled B fragments for k-steps s = chunk*8 .. chunk*8+7:
    // B[kl][o] = w2[s*1024 + kl*32 + o]; lane<16 holds kl=e2, lane>=16 holds kl=16+e2.
    for (int i = tid; i < 16 * 32 * 16; i += 128) {
      const int f = i >> 9, L = (i >> 4) & 31, e2 = i & 15;
      const int s = chunk * 8 + (f >> 1);
      const int t = f & 1;
      const int kl = e2 + ((L >= 16) ? 16 : 0);
      const int o  = t * 16 + (L & 15);
      w2s[f][L][e2] = (bf16_t)w2[s * 1024 + kl * 32 + o];
    }
    __syncthreads();
#pragma unroll
    for (int ss = 0; ss < 8; ++ss) {
      const int s = chunk * 8 + ss;
      const v16bf bf0 = *(const v16bf*)&w2s[ss * 2 + 0][lane][0];
      const v16bf bf1 = *(const v16bf*)&w2s[ss * 2 + 1][lane][0];
#pragma unroll
      for (int T = 0; T < 4; ++T) {
        const bf16_t hv = hsh[wave][T * 16 + m][s];
        const v16bf a = xf[T] * splat16(hv);  // rank-1 A fragment for this k-step
        acc[T][0] = __builtin_amdgcn_wmma_f32_16x16x32_bf16(false, a, false, bf0,
                                                            (short)0, acc[T][0], false, false);
        acc[T][1] = __builtin_amdgcn_wmma_f32_16x16x32_bf16(false, a, false, bf1,
                                                            (short)0, acc[T][1], false, false);
      }
    }
  }
  // en_b2 term: one extra step with A = x (h == 1)
  {
    const v16bf bb0 = *(const v16bf*)&b2s[0][lane][0];
    const v16bf bb1 = *(const v16bf*)&b2s[1][lane][0];
#pragma unroll
    for (int T = 0; T < 4; ++T) {
      acc[T][0] = __builtin_amdgcn_wmma_f32_16x16x32_bf16(false, xf[T], false, bb0,
                                                          (short)0, acc[T][0], false, false);
      acc[T][1] = __builtin_amdgcn_wmma_f32_16x16x32_bf16(false, xf[T], false, bb1,
                                                          (short)0, acc[T][1], false, false);
    }
  }
  // Scatter-add: C/D layout reg r -> row = r + 8*(lane>=16), col = t*16 + lane%16.
#pragma unroll
  for (int T = 0; T < 4; ++T) {
#pragma unroll
    for (int t = 0; t < 2; ++t) {
#pragma unroll
      for (int r = 0; r < 8; ++r) {
        const int row = r + (hihalf ? 8 : 0);
        const int eg  = ebase + T * 16 + row;
        if (eg < EE) {
          const int dn = dsts[wave][T * 16 + row];
          const int o  = t * 16 + (lane & 15);
          __hip_atomic_fetch_add(&agg[(long)dn * 32 + o], acc[T][t][r],
                                 __ATOMIC_RELAXED, __HIP_MEMORY_SCOPE_AGENT);
        }
      }
    }
  }
}

// ---------------------------------------------------------------------------
// GRU: m = relu(agg); h' = (1-z)*n + z*h   (fp32, out-of-place)
__global__ void gru_kernel(const float* __restrict__ agg, const float* __restrict__ hin,
                           float* __restrict__ hout, const float* __restrict__ wx,
                           const float* __restrict__ wh, const float* __restrict__ bx,
                           const float* __restrict__ bh) {
  int idx = blockIdx.x * blockDim.x + threadIdx.x;
  if (idx >= NN * 32) return;
  int n = idx >> 5, d = idx & 31;
  const float* mrow = agg + (long)n * 32;
  const float* hrow = hin + (long)n * 32;
  float ir = bx[d], iz = bx[32 + d], inn = bx[64 + d];
  float hr = bh[d], hz = bh[32 + d], hn = bh[64 + d];
#pragma unroll 4
  for (int i = 0; i < 32; ++i) {
    float mi = fmaxf(mrow[i], 0.0f);
    float hv = hrow[i];
    ir += mi * wx[i * 96 + d];  iz += mi * wx[i * 96 + 32 + d];  inn += mi * wx[i * 96 + 64 + d];
    hr += hv * wh[i * 96 + d];  hz += hv * wh[i * 96 + 32 + d];  hn  += hv * wh[i * 96 + 64 + d];
  }
  float r = sigf(ir + hr);
  float z = sigf(iz + hz);
  float nn = tanhf(inn + r * hn);
  hout[idx] = (1.0f - z) * nn + z * hrow[d];
}

// ---------------------------------------------------------------------------
// LSTM layer (torch gate order i,f,g,o); h out-of-place, c in-place
__global__ void lstm_kernel(const float* __restrict__ xin, int xdim,
                            const float* __restrict__ hin, float* __restrict__ hout,
                            float* __restrict__ c, const float* __restrict__ wx,
                            const float* __restrict__ wh, const float* __restrict__ bx,
                            const float* __restrict__ bh) {
  int idx = blockIdx.x * blockDim.x + threadIdx.x;
  if (idx >= BBG * 32) return;
  int b = idx >> 5, d = idx & 31;
  float gi = bx[d] + bh[d], gf = bx[32 + d] + bh[32 + d];
  float gg = bx[64 + d] + bh[64 + d], go = bx[96 + d] + bh[96 + d];
  for (int i = 0; i < xdim; ++i) {
    float x = xin[(long)b * xdim + i];
    gi += x * wx[i * 128 + d];       gf += x * wx[i * 128 + 32 + d];
    gg += x * wx[i * 128 + 64 + d];  go += x * wx[i * 128 + 96 + d];
  }
#pragma unroll 4
  for (int i = 0; i < 32; ++i) {
    float h = hin[(long)b * 32 + i];
    gi += h * wh[i * 128 + d];       gf += h * wh[i * 128 + 32 + d];
    gg += h * wh[i * 128 + 64 + d];  go += h * wh[i * 128 + 96 + d];
  }
  float c2 = sigf(gf) * c[idx] + sigf(gi) * tanhf(gg);
  c[idx] = c2;
  hout[idx] = sigf(go) * tanhf(c2);
}

// ---------------------------------------------------------------------------
// Segment offsets via binary search (graph_ids sorted ascending)
__global__ void segoff_kernel(const int* __restrict__ gid, int* __restrict__ seg) {
  int g = blockIdx.x * blockDim.x + threadIdx.x;
  if (g > BBG) return;
  if (g == BBG) { seg[g] = NN; return; }
  int lo = 0, hi = NN;
  while (lo < hi) { int mid = (lo + hi) >> 1; if (gid[mid] < g) lo = mid + 1; else hi = mid; }
  seg[g] = lo;
}

// ---------------------------------------------------------------------------
// Set2Set attention: block per graph. e=out.q, segment softmax, r = sum(alpha*out),
// q_star = [q | r]
__global__ __launch_bounds__(128) void s2s_attn_kernel(const float* __restrict__ out,
                                                       const float* __restrict__ q,
                                                       const int* __restrict__ seg,
                                                       float* __restrict__ qstar) {
  __shared__ float qsh[32];
  __shared__ float esh[256];
  __shared__ float red[128];
  const int g = blockIdx.x;
  const int tid = threadIdx.x;
  const int n0 = seg[g];
  int cnt = seg[g + 1] - n0;
  if (cnt > 256) cnt = 256;  // cannot happen for this input (~98 nodes/graph)
  if (tid < 32) qsh[tid] = q[(long)g * 32 + tid];
  __syncthreads();

  float lmax = -3.0e38f;
  for (int j = tid; j < cnt; j += 128) {
    const float* row = out + (long)(n0 + j) * 32;
    float e = 0.0f;
#pragma unroll
    for (int d = 0; d < 32; ++d) e += row[d] * qsh[d];
    esh[j] = e;
    lmax = fmaxf(lmax, e);
  }
  red[tid] = lmax;
  __syncthreads();
  for (int s = 64; s > 0; s >>= 1) {
    if (tid < s) red[tid] = fmaxf(red[tid], red[tid + s]);
    __syncthreads();
  }
  const float mx = red[0];
  __syncthreads();

  float lsum = 0.0f;
  for (int j = tid; j < cnt; j += 128) { float ex = expf(esh[j] - mx); esh[j] = ex; lsum += ex; }
  red[tid] = lsum;
  __syncthreads();
  for (int s = 64; s > 0; s >>= 1) {
    if (tid < s) red[tid] += red[tid + s];
    __syncthreads();
  }
  const float invS = 1.0f / red[0];
  __syncthreads();

  const int d = tid & 31, ch = tid >> 5;
  float pr = 0.0f;
  for (int j = ch; j < cnt; j += 4) pr += esh[j] * out[(long)(n0 + j) * 32 + d];
  red[tid] = pr * invS;
  __syncthreads();
  if (ch == 0) {
    float r = red[tid] + red[tid + 32] + red[tid + 64] + red[tid + 96];
    qstar[(long)g * 64 + d] = qsh[d];
    qstar[(long)g * 64 + 32 + d] = r;
  }
}

// ---------------------------------------------------------------------------
__global__ void lin1_kernel(const float* __restrict__ qstar, const float* __restrict__ w,
                            const float* __restrict__ b, float* __restrict__ y1) {
  int idx = blockIdx.x * blockDim.x + threadIdx.x;
  if (idx >= BBG * 32) return;
  int bb = idx >> 5, d = idx & 31;
  float acc = b[d];
#pragma unroll 8
  for (int i = 0; i < 64; ++i) acc += qstar[(long)bb * 64 + i] * w[i * 32 + d];
  y1[idx] = fmaxf(acc, 0.0f);
}

__global__ void lin2_kernel(const float* __restrict__ y1, const float* __restrict__ w,
                            const float* __restrict__ b, float* __restrict__ yout) {
  int idx = blockIdx.x * blockDim.x + threadIdx.x;
  if (idx >= BBG * 32) return;
  int bb = idx >> 5, o = idx & 31;
  float acc = b[o];
#pragma unroll 8
  for (int i = 0; i < 32; ++i) acc += y1[(long)bb * 32 + i] * w[i * 32 + o];
  yout[idx] = acc;
}

// ---------------------------------------------------------------------------
extern "C" void kernel_launch(void* const* d_in, const int* in_sizes, int n_in,
                              void* d_out, int out_size, void* d_ws, size_t ws_size,
                              hipStream_t stream) {
  (void)in_sizes; (void)n_in; (void)out_size; (void)ws_size;
  const float* pu     = (const float*)d_in[0];
  const float* pd     = (const float*)d_in[1];
  const float* seed   = (const float*)d_in[2];
  const float* nemb   = (const float*)d_in[3];
  const float* eemb   = (const float*)d_in[4];
  const float* lin0_w = (const float*)d_in[5];
  const float* lin0_b = (const float*)d_in[6];
  const float* en_w1  = (const float*)d_in[7];
  const float* en_b1  = (const float*)d_in[8];
  const float* en_w2  = (const float*)d_in[9];
  const float* en_b2  = (const float*)d_in[10];
  const float* conv_b = (const float*)d_in[11];
  const float* gru_wx = (const float*)d_in[12];
  const float* gru_wh = (const float*)d_in[13];
  const float* gru_bx = (const float*)d_in[14];
  const float* gru_bh = (const float*)d_in[15];
  const float* lstm_p[3][4];
  for (int l = 0; l < 3; ++l)
    for (int j = 0; j < 4; ++j) lstm_p[l][j] = (const float*)d_in[16 + 4 * l + j];
  const float* lin1_w = (const float*)d_in[28];
  const float* lin1_b = (const float*)d_in[29];
  const float* lin2_w = (const float*)d_in[30];
  const float* lin2_b = (const float*)d_in[31];
  const int* nfreq = (const int*)d_in[32];
  const int* efreq = (const int*)d_in[33];
  const int* srcI  = (const int*)d_in[34];
  const int* dstI  = (const int*)d_in[35];
  const int* gid   = (const int*)d_in[36];

  char* wsp = (char*)d_ws;
  auto alloc = [&](size_t bytes) -> void* {
    void* p = (void*)wsp;
    wsp += (bytes + 255) & ~(size_t)255;
    return p;
  };
  float* hA    = (float*)alloc((size_t)NN * 32 * 4);
  float* hB    = (float*)alloc((size_t)NN * 32 * 4);
  float* agg   = (float*)alloc((size_t)NN * 32 * 4);
  float* heb   = (float*)alloc((size_t)EE * 32 * 4);
  float* qstar = (float*)alloc((size_t)BBG * 64 * 4);
  float* y1    = (float*)alloc((size_t)BBG * 32 * 4);
  float* hbank = (float*)alloc((size_t)3 * 2 * BBG * 32 * 4);
  float* cbank = (float*)alloc((size_t)3 * BBG * 32 * 4);
  int*   seg   = (int*)alloc((size_t)(BBG + 1) * 4);
  const size_t B32 = (size_t)BBG * 32;

  segoff_kernel<<<(BBG + 1 + 255) / 256, 256, 0, stream>>>(gid, seg);
  lin0_kernel<<<(NN * 32 + 255) / 256, 256, 0, stream>>>(pu, pd, seed, nemb, lin0_w, lin0_b,
                                                         nfreq, hA);
  edge_hidden_kernel<<<(EE * 32 + 255) / 256, 256, 0, stream>>>(eemb, en_w1, en_b1, efreq, heb);

  for (int s = 0; s < 6; ++s) {
    float* cur = (s & 1) ? hB : hA;
    float* nxt = (s & 1) ? hA : hB;
    agg_init_kernel<<<(NN * 32 + 255) / 256, 256, 0, stream>>>(agg, conv_b);
    mp_edge_kernel<<<(EE + 255) / 256, 128, 0, stream>>>(cur, heb, en_w2, en_b2, srcI, dstI, agg);
    gru_kernel<<<(NN * 32 + 255) / 256, 256, 0, stream>>>(agg, cur, nxt, gru_wx, gru_wh,
                                                          gru_bx, gru_bh);
  }
  float* outN = hA;  // after 6 steps, final h lands back in hA

  hipMemsetAsync(qstar, 0, (size_t)BBG * 64 * 4, stream);
  hipMemsetAsync(hbank, 0, (size_t)3 * 2 * B32 * 4, stream);
  hipMemsetAsync(cbank, 0, (size_t)3 * B32 * 4, stream);

  for (int it = 0; it < 6; ++it) {
    int ib = it & 1, ob = ib ^ 1;
    lstm_kernel<<<(BBG * 32 + 255) / 256, 256, 0, stream>>>(
        qstar, 64, hbank + (0 * 2 + ib) * B32, hbank + (0 * 2 + ob) * B32, cbank + 0 * B32,
        lstm_p[0][0], lstm_p[0][1], lstm_p[0][2], lstm_p[0][3]);
    lstm_kernel<<<(BBG * 32 + 255) / 256, 256, 0, stream>>>(
        hbank + (0 * 2 + ob) * B32, 32, hbank + (1 * 2 + ib) * B32, hbank + (1 * 2 + ob) * B32,
        cbank + 1 * B32, lstm_p[1][0], lstm_p[1][1], lstm_p[1][2], lstm_p[1][3]);
    lstm_kernel<<<(BBG * 32 + 255) / 256, 256, 0, stream>>>(
        hbank + (1 * 2 + ob) * B32, 32, hbank + (2 * 2 + ib) * B32, hbank + (2 * 2 + ob) * B32,
        cbank + 2 * B32, lstm_p[2][0], lstm_p[2][1], lstm_p[2][2], lstm_p[2][3]);
    s2s_attn_kernel<<<BBG, 128, 0, stream>>>(outN, hbank + (2 * 2 + ob) * B32, seg, qstar);
  }

  lin1_kernel<<<(BBG * 32 + 255) / 256, 256, 0, stream>>>(qstar, lin1_w, lin1_b, y1);
  lin2_kernel<<<(BBG * 32 + 255) / 256, 256, 0, stream>>>(y1, lin2_w, lin2_b, (float*)d_out);
}
